// UnifiedFractalAttention_66425964200590
// MI455X (gfx1250) — compile-verified, matrix operand order checked by reading, not run on hardware
//
#include <hip/hip_runtime.h>
#include <hip/hip_bf16.h>

typedef __attribute__((ext_vector_type(16))) _Float16 v16h;
typedef __attribute__((ext_vector_type(8)))  _Float16 v8h;
typedef __attribute__((ext_vector_type(8)))  float    v8f;

#define BATCH        8
#define SEQ          4096
#define DIM          512
#define NDEPTH       8
#define PTOT         480          // sum of per-depth widths padded to mult. of 32
#define ROWS_PER_WG  64
#define WAVES_PER_WG 4
#define NTHREADS     (WAVES_PER_WG * 32)
#define QSTRIDE      520          // LDS row stride (halves), bank-conflict free
#define PSTRIDE      488          // LDS row stride (halves), bank-conflict free

// ---------------------------------------------------------------------------
// Setup: replicate numpy's double-precision Cantor construction exactly,
// dedupe indices (positions monotone -> adjacent-dedupe == np.unique),
// and compute softmax(scale_weights / scale_temperature).
// ---------------------------------------------------------------------------
__global__ void cantor_setup_kernel(const float* __restrict__ sw,
                                    const float* __restrict__ st,
                                    int* __restrict__ idxbuf,
                                    int* __restrict__ counts,
                                    float* __restrict__ wts)
{
    if (threadIdx.x != 0 || blockIdx.x != 0) return;

    const int off32[NDEPTH] = {0, 32, 64, 96, 128, 160, 224, 320};
    const int w32[NDEPTH]   = {32, 32, 32, 32, 32, 64, 96, 160};

    double prev[129], cur[129];
    prev[0] = 0.0; prev[1] = 1.0;
    int plen = 2;

    for (int d = 0; d < NDEPTH; ++d) {
        long long last = -1;
        int cnt = 0;
        for (int i = 0; i < plen; ++i) {
            long long iv = (long long)(prev[i] * 4095.0);  // trunc == np astype
            if (iv != last) { idxbuf[off32[d] + cnt] = (int)iv; ++cnt; last = iv; }
        }
        counts[d] = cnt;
        for (int i = cnt; i < w32[d]; ++i) idxbuf[off32[d] + i] = 0; // safe rows
        if (d + 1 < NDEPTH) {
            int n = 0;
            for (int i = 0; i < plen - 1; ++i) {
                double l = prev[i], r = prev[i + 1];
                double third = (r - l) / 3.0;
                cur[n++] = l;
                cur[n++] = l + third;
            }
            cur[n++] = prev[plen - 1];
            for (int i = 0; i < n; ++i) prev[i] = cur[i];
            plen = n;
        }
    }

    float t = st[0];
    float v[NDEPTH], mx = -3.4e38f, s = 0.f;
    for (int i = 0; i < NDEPTH; ++i) { v[i] = sw[i] / t; mx = fmaxf(mx, v[i]); }
    for (int i = 0; i < NDEPTH; ++i) { v[i] = __expf(v[i] - mx); s += v[i]; }
    for (int i = 0; i < NDEPTH; ++i) wts[i] = v[i] / s;
}

// ---------------------------------------------------------------------------
// Pre-gather K rows (f32 -> f16), compact slot-major:  Kg[b][slot][feature]
// ---------------------------------------------------------------------------
__global__ __launch_bounds__(256)
void gather_k_kernel(const float* __restrict__ K, const int* __restrict__ idx,
                     _Float16* __restrict__ Kg)
{
    const int b    = blockIdx.y;
    const int s    = blockIdx.x * 8 + (threadIdx.x >> 5);
    const int lane = threadIdx.x & 31;
    const float*    src = K  + ((size_t)(b * SEQ + idx[s])) * DIM + lane * 16;
    _Float16*       dst = Kg + ((size_t)(b * PTOT + s)) * DIM + lane * 16;
#pragma unroll
    for (int i = 0; i < 16; ++i) dst[i] = (_Float16)src[i];
}

// ---------------------------------------------------------------------------
// Pre-gather + transpose V (f32 -> f16):  Vt[b][feature][slot]
// ---------------------------------------------------------------------------
__global__ __launch_bounds__(256)
void gather_v_kernel(const float* __restrict__ V, const int* __restrict__ idx,
                     _Float16* __restrict__ Vt)
{
    __shared__ _Float16 T[32][516];
    const int b   = blockIdx.y;
    const int s0  = blockIdx.x * 32;
    const int tid = threadIdx.x;

    for (int i = tid; i < 32 * 512; i += 256) {
        const int s = i >> 9, f = i & 511;
        T[s][f] = (_Float16)V[((size_t)(b * SEQ + idx[s0 + s])) * DIM + f];
    }
    __syncthreads();
    for (int i = tid; i < 32 * 512; i += 256) {
        const int f = i >> 5, s = i & 31;
        Vt[((size_t)(b * DIM + f)) * PTOT + s0 + s] = T[s][f];
    }
}

// Fragment load (global or LDS f16): element e, lane-half h:
//   elems 0..7 -> K = 8h + 0..7 ; elems 8..15 -> K = 16 + 8h + 0..7
__device__ inline v16h load_frag_f16(const _Float16* p)
{
    const v8h lo = *(const v8h*)(p);
    const v8h hi = *(const v8h*)(p + 16);
    v16h r;
#pragma unroll
    for (int i = 0; i < 8; ++i) { r[i] = lo[i]; r[i + 8] = hi[i]; }
    return r;
}

// ---------------------------------------------------------------------------
// Fused Cantor attention. One workgroup = one batch x 64 query rows.
// Q tile staged once in LDS as f16; P (all depths' probabilities) in LDS.
// ---------------------------------------------------------------------------
__global__ __launch_bounds__(NTHREADS)
void cantor_attn_kernel(const float* __restrict__ Q,
                        const _Float16* __restrict__ Kg,
                        const _Float16* __restrict__ Vt,
                        const int* __restrict__ cntg,
                        const float* __restrict__ wtsg,
                        float* __restrict__ out)
{
    constexpr int off32[NDEPTH] = {0, 32, 64, 96, 128, 160, 224, 320};
    constexpr int w32[NDEPTH]   = {32, 32, 32, 32, 32, 64, 96, 160};

    __shared__ _Float16 Qh[ROWS_PER_WG][QSTRIDE];  // 66.6 KB
    __shared__ _Float16 P[ROWS_PER_WG][PSTRIDE];   // 62.5 KB
    __shared__ int      sCnt[NDEPTH];
    __shared__ float    sW[NDEPTH];

    const int tid  = threadIdx.x;
    const int lane = tid & 31;
    const int wv   = tid >> 5;
    const int lm   = lane & 15;
    const int lh   = lane >> 4;
    const int b    = blockIdx.y;
    const int q0   = blockIdx.x * ROWS_PER_WG;

    if (tid < NDEPTH) { sCnt[tid] = cntg[tid]; sW[tid] = wtsg[tid]; }

    // ---- Stage Q tile (f32 -> f16) into LDS, coalesced, once ----
    {
        const float* qbase = Q + ((size_t)(b * SEQ + q0)) * DIM;
        for (int i = tid; i < ROWS_PER_WG * DIM / 4; i += NTHREADS) {
            const int flat = i * 4;
            const int r = flat >> 9;        // /512
            const int c = flat & 511;
            const float4 x = *(const float4*)(qbase + (size_t)r * DIM + c);
            Qh[r][c + 0] = (_Float16)x.x;
            Qh[r][c + 1] = (_Float16)x.y;
            Qh[r][c + 2] = (_Float16)x.z;
            Qh[r][c + 3] = (_Float16)x.w;
        }
    }
    __syncthreads();

    const float inv_scale = 0.04419417382415922f;  // 1/sqrt(512)

    // ---------------- Phase 1: scores ----------------
    const _Float16* kgb = Kg + ((size_t)b * PTOT) * DIM + lh * 8;
#pragma unroll
    for (int d = 0; d < NDEPTH; ++d) {
        const int ntiles = w32[d] >> 4;           // constant after unroll
        v8f c[10];
        const _Float16* kp[10];
#pragma unroll
        for (int t = 0; t < ntiles; ++t) {
            c[t]  = v8f{0.f,0.f,0.f,0.f,0.f,0.f,0.f,0.f};
            kp[t] = kgb + (size_t)(off32[d] + t * 16 + lm) * DIM;
        }
        for (int kc = 0; kc < DIM / 32; ++kc) {
            v16h a = load_frag_f16(&Qh[wv * 16 + lm][kc * 32 + lh * 8]);
#pragma unroll
            for (int t = 0; t < ntiles; ++t) {
                v16h bf = load_frag_f16(kp[t] + kc * 32);
                c[t] = __builtin_amdgcn_wmma_f32_16x16x32_f16(
                           false, a, false, bf, (short)0, c[t], false, false);
            }
        }
        const int cnt = sCnt[d];
#pragma unroll
        for (int t = 0; t < ntiles; ++t) {
            const bool valid = (t * 16 + lm) < cnt;
            const int  col   = off32[d] + t * 16 + lm;
#pragma unroll
            for (int r = 0; r < 8; ++r) {
                float s = valid ? c[t][r] * inv_scale : -__builtin_inff();
                P[wv * 16 + r + lh * 8][col] = (_Float16)s;
            }
        }
    }
    __syncthreads();

    // ---------------- Softmax (depth weight folded in) ----------------
    for (int task = tid; task < ROWS_PER_WG * NDEPTH; task += NTHREADS) {
        const int r = task & (ROWS_PER_WG - 1);
        const int d = task >> 6;
        const int o = off32[d], wdt = w32[d], cnt = sCnt[d];
        float mx = -__builtin_inff();
        for (int j = 0; j < cnt; ++j) mx = fmaxf(mx, (float)P[r][o + j]);
        float sum = 0.f;
        for (int j = 0; j < cnt; ++j) sum += __expf((float)P[r][o + j] - mx);
        const float sc = sW[d] / sum;
        for (int j = 0; j < wdt; ++j) {   // padded cols: exp(-inf)=0 -> P=0
            float x = (float)P[r][o + j];
            P[r][o + j] = (_Float16)(__expf(x - mx) * sc);
        }
    }
    __syncthreads();

    // ---------------- Phase 2: out = sum_d w_d * P_d @ V_d ----------------
    const size_t    orow0 = (size_t)(b * SEQ + q0 + wv * 16);
    const _Float16* vtb   = Vt + ((size_t)b * DIM) * PTOT + lh * 8;
#pragma unroll 1
    for (int nb = 0; nb < 4; ++nb) {              // 4 x 128 output columns
        v8f o[8];
        const _Float16* vp[8];
#pragma unroll
        for (int nt = 0; nt < 8; ++nt) {
            o[nt]  = v8f{0.f,0.f,0.f,0.f,0.f,0.f,0.f,0.f};
            vp[nt] = vtb + (size_t)(nb * 128 + nt * 16 + lm) * PTOT;
        }
        for (int kc = 0; kc < PTOT / 32; ++kc) {  // all depths, padded P==0
            v16h a = load_frag_f16(&P[wv * 16 + lm][kc * 32 + lh * 8]);
#pragma unroll
            for (int nt = 0; nt < 8; ++nt) {
                v16h bf = load_frag_f16(vp[nt] + kc * 32);
                o[nt] = __builtin_amdgcn_wmma_f32_16x16x32_f16(
                            false, a, false, bf, (short)0, o[nt], false, false);
            }
        }
#pragma unroll
        for (int nt = 0; nt < 8; ++nt) {
#pragma unroll
            for (int r = 0; r < 8; ++r) {
                out[(orow0 + r + lh * 8) * DIM + nb * 128 + nt * 16 + lm] = o[nt][r];
            }
        }
    }
}

// ---------------------------------------------------------------------------
extern "C" void kernel_launch(void* const* d_in, const int* in_sizes, int n_in,
                              void* d_out, int out_size, void* d_ws, size_t ws_size,
                              hipStream_t stream)
{
    const float* Q  = (const float*)d_in[0];
    const float* K  = (const float*)d_in[1];
    const float* V  = (const float*)d_in[2];
    // d_in[3] = t  (unused by the reference computation)
    const float* sw = (const float*)d_in[4];
    const float* st = (const float*)d_in[5];

    char* ws = (char*)d_ws;
    int*   idxbuf = (int*)ws;                         // PTOT ints
    int*   counts = idxbuf + PTOT;                    // NDEPTH ints
    float* wts    = (float*)(counts + NDEPTH);        // NDEPTH floats

    _Float16* Kg = (_Float16*)(ws + 4096);                       // [B][PTOT][DIM]
    _Float16* Vt = Kg + (size_t)BATCH * PTOT * DIM;              // [B][DIM][PTOT]

    hipLaunchKernelGGL(cantor_setup_kernel, dim3(1), dim3(1), 0, stream,
                       sw, st, idxbuf, counts, wts);

    hipLaunchKernelGGL(gather_k_kernel, dim3(PTOT / 8, BATCH), dim3(256), 0, stream,
                       K, idxbuf, Kg);
    hipLaunchKernelGGL(gather_v_kernel, dim3(PTOT / 32, BATCH), dim3(256), 0, stream,
                       V, idxbuf, Vt);

    hipLaunchKernelGGL(cantor_attn_kernel, dim3(SEQ / ROWS_PER_WG, BATCH),
                       dim3(NTHREADS), 0, stream,
                       Q, Kg, Vt, counts, wts, (float*)d_out);
}
